// RotaryPositionMultiHeadedAttention_48911087567015
// MI455X (gfx1250) — matrix operand order repaired and checked by reading, not compiled, not run on hardware
//
#include <hip/hip_runtime.h>
#include <hip/hip_bf16.h>
#include <math.h>

typedef __attribute__((ext_vector_type(16))) _Float16 v16h;
typedef __attribute__((ext_vector_type(8)))  _Float16 v8h;
typedef __attribute__((ext_vector_type(2)))  _Float16 v2h;
typedef __attribute__((ext_vector_type(8)))  float    v8f;
typedef int v4i __attribute__((vector_size(16)));

#define TT  2048
#define BB  2
#define CC  1024
#define HH  16
#define DKK 64

// ---------- fragment loaders (per CDNA5 ISA §7.12.2 VGPR layouts, wave32) ----------

__device__ __forceinline__ v8h ld8(const _Float16* p) { return *(const v8h*)p; }

__device__ __forceinline__ v16h cat16(v8h lo, v8h hi) {
  v16h r;
  #pragma unroll
  for (int i = 0; i < 8; ++i) { r[i] = lo[i]; r[i + 8] = hi[i]; }
  return r;
}

// A-fragment 16x32 f16 (MxK). base = &A[m_base][k0], row stride lda (elements).
// Lane l: m = l&15, hi = l>>4. halves 0..7 -> K = 8*hi+0..7 ; halves 8..15 -> K = 16+8*hi+0..7
__device__ __forceinline__ v16h ld_afrag(const _Float16* base, int lda, int lane) {
  int m = lane & 15, hi = lane >> 4;
  const _Float16* p = base + m * lda + 8 * hi;
  return cat16(ld8(p), ld8(p + 16));
}

// B-fragment 32x16 f16 (KxN). Source: element (k,n) at base[n*ldb + k] (column n K-contiguous).
// Lane l: n = l&15, hi = l>>4. halves 0..7 -> K = 8*hi+0..7 ; halves 8..15 -> K = 16+8*hi+0..7
__device__ __forceinline__ v16h ld_bfrag(const _Float16* base, int ldb, int lane) {
  int n = lane & 15, hi = lane >> 4;
  const _Float16* p = base + n * ldb + 8 * hi;
  return cat16(ld8(p), ld8(p + 16));
}

// async 16B global -> LDS copy (ASYNCcnt path); builtin takes v4i pointers
__device__ __forceinline__ void async_cp16(const _Float16* g, _Float16* l) {
  __builtin_amdgcn_global_load_async_to_lds_b128(
      (__attribute__((address_space(1))) v4i*)(g),
      (__attribute__((address_space(3))) v4i*)(l),
      0, 0);
}

// ---------- weight f32 -> f16 ----------
__global__ void __launch_bounds__(256)
cvt_w_kernel(const float* __restrict__ Wq, const float* __restrict__ Wk,
             const float* __restrict__ Wv, const float* __restrict__ Wo,
             _Float16* __restrict__ Wh) {
  int idx = blockIdx.x * 256 + threadIdx.x;          // < 4M
  int sel = idx >> 20;
  int off = idx & ((1 << 20) - 1);
  const float* s = (sel == 0) ? Wq : (sel == 1) ? Wk : (sel == 2) ? Wv : Wo;
  Wh[idx] = (_Float16)s[off];
}

// ---------- RoPE + [T,B,C]->[B,T,C] + f16 cast ----------
__global__ void __launch_bounds__(256)
rope_pack_kernel(const float* __restrict__ q, const float* __restrict__ k,
                 const float* __restrict__ v, _Float16* __restrict__ qr,
                 _Float16* __restrict__ kr, _Float16* __restrict__ vr) {
  int idx = blockIdx.x * 256 + threadIdx.x;          // < T*B*C = 4M
  int c   = idx & (CC - 1);
  int rem = idx >> 10;
  int b   = rem & (BB - 1);
  int t   = rem >> 1;
  int d   = c & (DKK - 1);
  int j   = d & 31;
  // inv_freq = 10000^(-j/32) = exp2(-j * log2(10000)/32)
  float inv = exp2f(-0.41524101186092f * (float)j);
  float ang = (float)t * inv;
  float sv, cv;
  sincosf(ang, &sv, &cv);
  float qv = q[idx], kv = k[idx];
  int   pidx = (d < 32) ? idx + 32 : idx - 32;       // rotate_half partner
  float sgn  = (d < 32) ? -1.f : 1.f;
  float qp = q[pidx] * sgn, kp = k[pidx] * sgn;
  size_t o = ((size_t)(b * TT + t)) * CC + c;
  qr[o] = (_Float16)(qv * cv + qp * sv);
  kr[o] = (_Float16)(kv * cv + kp * sv);
  vr[o] = (_Float16)v[idx];
}

// ---------- WMMA GEMM core with async-LDS staged W tiles ----------
// Block = 256 threads (8 waves). Each wave: 16 rows. All waves share W columns
// [n_base, n_base+64). W tile 64x64 f16 (8 KB) double-buffered in LDS, filled
// with global_load_async_to_lds_b128 and consumed via ds_load_b128 B-fragments.
__device__ __forceinline__ void gemm_core_staged(const _Float16* __restrict__ X,
                                                 const _Float16* __restrict__ W,
                                                 _Float16* wtile,   // [2][64*64] LDS
                                                 int m_base, int n_base,
                                                 int lane, v8f acc[4]) {
  const int tid = threadIdx.x;
  #pragma unroll
  for (int j = 0; j < 4; ++j)
    #pragma unroll
    for (int r = 0; r < 8; ++r) acc[j][r] = 0.f;

  // stage W[n_base+row][kk + seg*8 ..+7] -> wtile[buf][row*64 + seg*8]
  auto stage = [&](int buf, int kk) {
    #pragma unroll
    for (int i = 0; i < 2; ++i) {
      int idx = tid + i * 256;                       // 512 x 16B = 8 KB
      int row = idx >> 3, seg = idx & 7;
      async_cp16(W + (size_t)(n_base + row) * CC + kk + seg * 8,
                 wtile + buf * (64 * 64) + row * 64 + seg * 8);
    }
  };

  stage(0, 0);
  for (int kk = 0; kk < CC; kk += 64) {
    int buf = (kk >> 6) & 1;
    asm volatile("s_wait_asynccnt 0" ::: "memory");  // own async loads done
    __syncthreads();                                  // whole tile visible
    if (kk + 64 < CC) stage(buf ^ 1, kk + 64);        // prefetch next tile
    #pragma unroll
    for (int kc = 0; kc < 64; kc += 32) {
      __builtin_prefetch(X + (size_t)m_base * CC + kk + 128, 0, 1);
      v16h a = ld_afrag(X + (size_t)m_base * CC + kk + kc, CC, lane);
      #pragma unroll
      for (int j = 0; j < 4; ++j) {
        v16h bf = ld_bfrag(wtile + buf * (64 * 64) + (j * 16) * 64 + kc, 64, lane);
        acc[j] = __builtin_amdgcn_wmma_f32_16x16x32_f16(false, a, false, bf,
                                                        (short)0, acc[j], false, false);
      }
    }
    __syncthreads();                                  // reads done before refill
  }
}

// ---------- QKV projection: Y = X*W^T + b ; vmode 0 -> [B,H,T,DK], 1 -> [B,H,DK,T] ----------
__global__ void __launch_bounds__(256)
gemm_qkv_kernel(const _Float16* __restrict__ X, const _Float16* __restrict__ W,
                const float* __restrict__ bias, _Float16* __restrict__ Y, int vmode) {
  __shared__ _Float16 wtile[2 * 64 * 64];
  int warp = threadIdx.x >> 5, lane = threadIdx.x & 31;
  int mt = blockIdx.x * 8 + warp;
  int m_base = mt * 16, n_base = blockIdx.y * 64;
  v8f acc[4];
  gemm_core_staged(X, W, wtile, m_base, n_base, lane, acc);
  int n0 = n_base + (lane & 15);
  int hi = lane >> 4;
  #pragma unroll
  for (int j = 0; j < 4; ++j) {
    int   nn = n0 + j * 16;
    float bj = bias[nn];
    int   h = nn >> 6, d = nn & (DKK - 1);
    #pragma unroll
    for (int r = 0; r < 8; ++r) {
      int m = m_base + r + 8 * hi;
      int b = m >> 11, t = m & (TT - 1);
      size_t addr = vmode ? ((size_t)((b * HH + h) * DKK + d)) * TT + t
                          : ((size_t)((b * HH + h) * TT + t)) * DKK + d;
      Y[addr] = (_Float16)(acc[j][r] + bj);
    }
  }
}

// ---------- output projection: f32 result into [T,B,C] ----------
__global__ void __launch_bounds__(256)
gemm_out_kernel(const _Float16* __restrict__ X, const _Float16* __restrict__ W,
                const float* __restrict__ bias, float* __restrict__ Y) {
  __shared__ _Float16 wtile[2 * 64 * 64];
  int warp = threadIdx.x >> 5, lane = threadIdx.x & 31;
  int mt = blockIdx.x * 8 + warp;
  int m_base = mt * 16, n_base = blockIdx.y * 64;
  v8f acc[4];
  gemm_core_staged(X, W, wtile, m_base, n_base, lane, acc);
  int n0 = n_base + (lane & 15);
  int hi = lane >> 4;
  #pragma unroll
  for (int j = 0; j < 4; ++j) {
    int   nn = n0 + j * 16;
    float bj = bias[nn];
    #pragma unroll
    for (int r = 0; r < 8; ++r) {
      int m = m_base + r + 8 * hi;
      int b = m >> 11, t = m & (TT - 1);
      Y[((size_t)t * BB + b) * CC + nn] = acc[j][r] + bj;
    }
  }
}

// ---------- flash attention: one wave per (b,h, 16-query tile) ----------
__global__ void __launch_bounds__(128)
attn_kernel(const _Float16* __restrict__ Q, const _Float16* __restrict__ K,
            const _Float16* __restrict__ Vt, _Float16* __restrict__ O) {
  __shared__ _Float16 plds[4 * 16 * 32];             // per-wave 16x32 P staging
  const int warp = threadIdx.x >> 5;
  const int lane = threadIdx.x & 31;
  const int wid  = blockIdx.x * 4 + warp;            // 0..4095
  const int bh   = wid >> 7;                         // 0..31
  const int qt   = wid & 127;                        // query tile
  const int b    = bh >> 4;
  const int h    = bh & 15;
  const _Float16* Qh = Q  + (size_t)bh * TT * DKK;   // [T,DK]
  const _Float16* Kh = K  + (size_t)bh * TT * DKK;   // [T,DK]
  const _Float16* Vh = Vt + (size_t)bh * DKK * TT;   // [DK,T]
  _Float16* lp = plds + warp * (16 * 32);

  const int n  = lane & 15;
  const int hi = lane >> 4;

  // Q tile 16x64 as two A fragments (K-chunks 0..31, 32..63)
  v16h aq0 = ld_afrag(Qh + qt * 16 * DKK + 0,  DKK, lane);
  v16h aq1 = ld_afrag(Qh + qt * 16 * DKK + 32, DKK, lane);

  v8f   acc[4];
  float m_run[8], l_run[8];
  #pragma unroll
  for (int j = 0; j < 4; ++j)
    #pragma unroll
    for (int r = 0; r < 8; ++r) acc[j][r] = 0.f;
  #pragma unroll
  for (int r = 0; r < 8; ++r) { m_run[r] = -1e30f; l_run[r] = 0.f; }

  for (int kb = 0; kb < TT; kb += 32) {
    // S = Q K^T for 32 keys: two 16x16 tiles (keys kb..+15, kb+16..+31)
    v8f Sa, Sb;
    #pragma unroll
    for (int r = 0; r < 8; ++r) { Sa[r] = 0.f; Sb[r] = 0.f; }
    Sa = __builtin_amdgcn_wmma_f32_16x16x32_f16(false, aq0, false,
           ld_bfrag(Kh + (size_t)(kb +  0) * DKK +  0, DKK, lane), (short)0, Sa, false, false);
    Sa = __builtin_amdgcn_wmma_f32_16x16x32_f16(false, aq1, false,
           ld_bfrag(Kh + (size_t)(kb +  0) * DKK + 32, DKK, lane), (short)0, Sa, false, false);
    Sb = __builtin_amdgcn_wmma_f32_16x16x32_f16(false, aq0, false,
           ld_bfrag(Kh + (size_t)(kb + 16) * DKK +  0, DKK, lane), (short)0, Sb, false, false);
    Sb = __builtin_amdgcn_wmma_f32_16x16x32_f16(false, aq1, false,
           ld_bfrag(Kh + (size_t)(kb + 16) * DKK + 32, DKK, lane), (short)0, Sb, false, false);

    // online softmax; row m = r + 8*hi lives in one 16-lane half-group
    float cr[8];
    #pragma unroll
    for (int r = 0; r < 8; ++r) {
      float sa = Sa[r] * 0.125f;                     // 1/sqrt(64)
      float sb = Sb[r] * 0.125f;
      float mx = fmaxf(sa, sb);
      mx = fmaxf(mx, __shfl_xor(mx, 1));
      mx = fmaxf(mx, __shfl_xor(mx, 2));
      mx = fmaxf(mx, __shfl_xor(mx, 4));
      mx = fmaxf(mx, __shfl_xor(mx, 8));
      float mn = fmaxf(m_run[r], mx);
      float c  = __expf(m_run[r] - mn);
      float ea = __expf(sa - mn);
      float eb = __expf(sb - mn);
      float rs = ea + eb;
      rs += __shfl_xor(rs, 1);
      rs += __shfl_xor(rs, 2);
      rs += __shfl_xor(rs, 4);
      rs += __shfl_xor(rs, 8);
      l_run[r] = l_run[r] * c + rs;
      m_run[r] = mn;
      cr[r] = c;
      int row = r + 8 * hi;                          // D layout -> LDS row-major 16x32
      lp[row * 32 + n]      = (_Float16)ea;
      lp[row * 32 + n + 16] = (_Float16)eb;
    }
    #pragma unroll
    for (int j = 0; j < 4; ++j)
      #pragma unroll
      for (int r = 0; r < 8; ++r) acc[j][r] *= cr[r];

    asm volatile("s_wait_dscnt 0" ::: "memory");

    // gather P as 16x32 A fragment (documented 16-bit A swizzle), pairs -> ds_load_b32
    v16h ap;
    #pragma unroll
    for (int g = 0; g < 8; ++g) {
      int k0 = (g < 4) ? (8 * hi + 2 * g) : (16 + 8 * hi + 2 * (g - 4));
      v2h pr = *(const v2h*)(lp + n * 32 + k0);      // row m = lane&15
      ap[2 * g]     = pr[0];
      ap[2 * g + 1] = pr[1];
    }

    // O += P * V ; V^T layout gives K-contiguous B columns
    #pragma unroll
    for (int j = 0; j < 4; ++j) {
      acc[j] = __builtin_amdgcn_wmma_f32_16x16x32_f16(false, ap, false,
                 ld_bfrag(Vh + (size_t)(j * 16) * TT + kb, TT, lane),
                 (short)0, acc[j], false, false);
    }
  }

  float inv[8];
  #pragma unroll
  for (int r = 0; r < 8; ++r) inv[r] = 1.f / l_run[r];
  #pragma unroll
  for (int j = 0; j < 4; ++j) {
    #pragma unroll
    for (int r = 0; r < 8; ++r) {
      int t = qt * 16 + r + 8 * hi;
      int c = h * DKK + j * 16 + n;
      O[((size_t)(b * TT + t)) * CC + c] = (_Float16)(acc[j][r] * inv[r]);
    }
  }
}

// ---------- host-side launch ----------
extern "C" void kernel_launch(void* const* d_in, const int* in_sizes, int n_in,
                              void* d_out, int out_size, void* d_ws, size_t ws_size,
                              hipStream_t stream) {
  (void)in_sizes; (void)n_in; (void)out_size; (void)ws_size;
  const float* q  = (const float*)d_in[0];
  const float* k  = (const float*)d_in[1];
  const float* v  = (const float*)d_in[2];
  const float* Wq = (const float*)d_in[3];
  const float* bq = (const float*)d_in[4];
  const float* Wk = (const float*)d_in[5];
  const float* bk = (const float*)d_in[6];
  const float* Wv = (const float*)d_in[7];
  const float* bv = (const float*)d_in[8];
  const float* Wo = (const float*)d_in[9];
  const float* bo = (const float*)d_in[10];

  _Float16* ws = (_Float16*)d_ws;
  const size_t SEG = (size_t)1 << 22;                // 4M f16 = 8 MB; total 64 MB
  _Float16* Wh  = ws + 0 * SEG;                      // [4][C][C] f16
  _Float16* qr  = ws + 1 * SEG;                      // [B][T][C]
  _Float16* kr  = ws + 2 * SEG;
  _Float16* vr  = ws + 3 * SEG;
  _Float16* Qb  = ws + 4 * SEG;                      // [B][H][T][DK]
  _Float16* Kb  = ws + 5 * SEG;                      // [B][H][T][DK]
  _Float16* Vtb = ws + 6 * SEG;                      // [B][H][DK][T]
  _Float16* Ob  = ws + 7 * SEG;                      // [B][T][C]

  cvt_w_kernel<<<(4u << 20) / 256, 256, 0, stream>>>(Wq, Wk, Wv, Wo, Wh);
  rope_pack_kernel<<<(4u << 20) / 256, 256, 0, stream>>>(q, k, v, qr, kr, vr);
  dim3 gg(32, 16);                                   // (M/128, N/64), 8 waves/block
  gemm_qkv_kernel<<<gg, 256, 0, stream>>>(qr, Wh + 0 * (1 << 20), bq, Qb, 0);
  gemm_qkv_kernel<<<gg, 256, 0, stream>>>(kr, Wh + 1 * (1 << 20), bk, Kb, 0);
  gemm_qkv_kernel<<<gg, 256, 0, stream>>>(vr, Wh + 2 * (1 << 20), bv, Vtb, 1);
  attn_kernel<<<1024, 128, 0, stream>>>(Qb, Kb, Vtb, Ob);
  gemm_out_kernel<<<gg, 256, 0, stream>>>(Ob, Wh + 3 * (1 << 20), bo, (float*)d_out);
}
